// Tokenizer_30897994728273
// MI455X (gfx1250) — matrix operand order repaired and checked by reading, not compiled, not run on hardware
//
#include <hip/hip_runtime.h>
#include <stdint.h>

// ---------------------------------------------------------------------------
// VQ tokenizer forward for MI455X (gfx1250, wave32, WMMA bf16)
//   B=64, C=256, H=W=16, E=256, V=8192 -> NPIX = 16384
// ---------------------------------------------------------------------------
#define NPIX 16384
#define CDIM 256
#define EDIM 256
#define VDIM 8192
#define HWD  256
#define NREC 4194304   // B*C*H*W

#if defined(__gfx1250__) && __has_builtin(__builtin_amdgcn_global_load_async_to_lds_b128)
#define ASYNC_LDS 1
#pragma message("CDNA5 probe: __builtin_amdgcn_global_load_async_to_lds_b128 AVAILABLE")
#else
#define ASYNC_LDS 0
#pragma message("CDNA5 probe: __builtin_amdgcn_global_load_async_to_lds_b128 NOT available; plain LDS copy")
#endif

typedef __attribute__((ext_vector_type(16))) __bf16 v16bf;
typedef __attribute__((ext_vector_type(8)))  float  v8f;
typedef __attribute__((ext_vector_type(4)))  int    v4i;

union Frag { v16bf v; unsigned u[8]; uint4 q[2]; };

__device__ __forceinline__ unsigned bf16pk(float lo, float hi) {
  unsigned a = __float_as_uint(lo);
  unsigned b = __float_as_uint(hi);
  a = (a + 0x7FFFu + ((a >> 16) & 1u)) >> 16;          // rne
  b = (b + 0x7FFFu + ((b >> 16) & 1u)) & 0xFFFF0000u;
  return a | b;
}
__device__ __forceinline__ unsigned short bf16c(float x) {
  unsigned a = __float_as_uint(x);
  return (unsigned short)((a + 0x7FFFu + ((a >> 16) & 1u)) >> 16);
}

__device__ __forceinline__ v8f wmma_bf16(const Frag& A, const Frag& B, v8f C) {
  return __builtin_amdgcn_wmma_f32_16x16x32_bf16(false, A.v, false, B.v,
                                                 (short)0, C, false, false);
}

// ---------------------------------------------------------------------------
// Kernel 1: codebook f32 -> bf16 (row-major) + per-row sum of squares.
// One wave per row; 1024 blocks x 256 threads.
// ---------------------------------------------------------------------------
__global__ void k_convert_codebook(const float* __restrict__ cb,
                                   unsigned short* __restrict__ cbbf,
                                   float* __restrict__ cbssq) {
  const int lane = threadIdx.x & 31;
  const int wave = threadIdx.x >> 5;
  const int row  = blockIdx.x * 8 + wave;
  const float* src = cb + (size_t)row * EDIM;
  float ssq = 0.f;
  #pragma unroll
  for (int i = 0; i < 2; ++i) {
    float4 f = *(const float4*)(src + lane * 4 + i * 128);
    ssq += f.x * f.x + f.y * f.y + f.z * f.z + f.w * f.w;
    uint2 p;
    p.x = bf16pk(f.x, f.y);
    p.y = bf16pk(f.z, f.w);
    *(uint2*)(cbbf + (size_t)row * EDIM + lane * 4 + i * 128) = p;
  }
  #pragma unroll
  for (int m = 16; m >= 1; m >>= 1) ssq += __shfl_xor(ssq, m, 32);
  if (lane == 0) cbssq[row] = ssq;
}

// ---------------------------------------------------------------------------
// Kernel 2: pre-quant GEMM  z[p,e] = sum_c x[p,c]*W_pre[e,c] + b_pre[e]
// Block: 16 pixels x 256 E (wave w handles e-tile [32w,32w+32)); K loop over C.
// x tile staged through LDS as packed bf16 pairs in A-fragment layout.
// ---------------------------------------------------------------------------
__global__ void k_prequant(const float* __restrict__ x,
                           const float* __restrict__ Wpre,
                           const float* __restrict__ bpre,
                           unsigned short* __restrict__ zbf) {
  __shared__ unsigned xs[16][16];   // [kpair][m] packed bf16 pairs
  const int tid  = threadIdx.x;
  const int lane = tid & 31;
  const int wave = tid >> 5;
  const int hi   = lane >> 4;
  const int mn   = lane & 15;
  const int p0   = blockIdx.x * 16;
  const int bb   = p0 >> 8;        // batch
  const int hw0  = p0 & 255;       // h*W+w base
  const int e0   = wave * 32;
  const int kp_st = tid >> 4;      // staging kpair 0..15
  const int m_st  = tid & 15;      // staging pixel

  v8f acc0{}; v8f acc1{};
  for (int k0 = 0; k0 < CDIM; k0 += 32) {
    __syncthreads();
    // x[(b, c, hw)] : stride over c is HWD floats; consecutive m contiguous.
    const float* xp = x + (size_t)bb * CDIM * HWD + (size_t)(k0 + 2 * kp_st) * HWD + hw0 + m_st;
    xs[kp_st][m_st] = bf16pk(xp[0], xp[HWD]);
    __syncthreads();

    Frag A;
    #pragma unroll
    for (int j = 0; j < 8; ++j) {
      // A layout: lanes<16 K={0..7,16..23}, lanes>=16 K={8..15,24..31}
      int kp = (j & 3) + ((j >= 4) ? 8 : 0) + 4 * hi;
      A.u[j] = xs[kp][mn];
    }
    #pragma unroll
    for (int t = 0; t < 2; ++t) {
      // B[k,n] = W_pre[e0+16t+n, k0+k]; row contiguous in c.
      const float* wrow = Wpre + (size_t)(e0 + 16 * t + mn) * CDIM + k0 + 8 * hi;
      Frag Bf;
      #pragma unroll
      for (int j = 0; j < 8; ++j) {
        float2 f = *(const float2*)(wrow + 2 * (j & 3) + ((j >= 4) ? 16 : 0));
        Bf.u[j] = bf16pk(f.x, f.y);
      }
      if (t == 0) acc0 = wmma_bf16(A, Bf, acc0);
      else        acc1 = wmma_bf16(A, Bf, acc1);
    }
  }
  // D layout: lane holds column n; VGPR r -> row r (lanes<16) / r+8 (lanes>=16)
  #pragma unroll
  for (int t = 0; t < 2; ++t) {
    int e = e0 + 16 * t + mn;
    float bias = bpre[e];
    #pragma unroll
    for (int r = 0; r < 8; ++r) {
      int m = r + 8 * hi;
      float val = ((t == 0) ? acc0[r] : acc1[r]) + bias;
      zbf[(size_t)(p0 + m) * EDIM + e] = bf16c(val);
    }
  }
}

// ---------------------------------------------------------------------------
// Kernel 3: distances + argmin.  token = argmin_v (||cb_v||^2 - 2 z.cb_v)
// (||z||^2 dropped: constant per row).  Block: 16 pixels; each wave covers
// disjoint 16-v tiles, 64 iterations over V=8192.  A kept in registers.
// ---------------------------------------------------------------------------
__global__ void k_distargmin(const unsigned short* __restrict__ zbf,
                             const unsigned short* __restrict__ cbbf,
                             const float* __restrict__ cbssq,
                             int* __restrict__ tok,
                             float* __restrict__ outTok) {
  __shared__ __align__(16) unsigned short zsh[16 * EDIM];   // 8KB z tile
  __shared__ float rminS[8][16];
  __shared__ int   ridxS[8][16];

  const int tid  = threadIdx.x;
  const int lane = tid & 31;
  const int wave = tid >> 5;
  const int hi   = lane >> 4;
  const int n    = lane & 15;
  const int p0   = blockIdx.x * 16;

  // ---- stage z tile (16 rows x 512B) into LDS ----
#if ASYNC_LDS
  {
    // builtin prototype (per hipcc diagnostic): (v4i AS1*, v4i AS3*, imm, imm)
    typedef __attribute__((address_space(1))) v4i* gp_t;   // ptr to v4i in global
    typedef __attribute__((address_space(3))) v4i* lp_t;   // ptr to v4i in LDS
    #pragma unroll
    for (int i = 0; i < 2; ++i) {
      unsigned boff = (unsigned)(tid + 256 * i) * 16u;
      const char* g = (const char*)zbf + (size_t)p0 * EDIM * 2 + boff;
      char* l = (char*)zsh + boff;
      // AS1 value == flat global address; AS3 value == low 32 bits of flat LDS addr
      gp_t gp = (gp_t)(uintptr_t)g;
      lp_t lp = (lp_t)(uint32_t)(uintptr_t)l;
      __builtin_amdgcn_global_load_async_to_lds_b128(gp, lp, 0, 0);
    }
#if __has_builtin(__builtin_amdgcn_s_wait_asynccnt)
    __builtin_amdgcn_s_wait_asynccnt(0);
#else
    asm volatile("s_wait_asynccnt 0" ::: "memory");
#endif
  }
#else
  #pragma unroll
  for (int i = 0; i < 2; ++i) {
    int off = (tid + 256 * i) * 8;   // in shorts, 16B chunks
    *(uint4*)&zsh[off] = *(const uint4*)(zbf + (size_t)p0 * EDIM + off);
  }
#endif
  __syncthreads();

  // ---- build A fragments for all 8 K-steps, kept in VGPRs ----
  Frag A[8];
  #pragma unroll
  for (int s = 0; s < 8; ++s) {
    const unsigned short* zr = &zsh[n * EDIM + 32 * s + 8 * hi];
    A[s].q[0] = *(const uint4*)(zr);
    A[s].q[1] = *(const uint4*)(zr + 16);
  }

  float best[8]; int bidx[8];
  #pragma unroll
  for (int r = 0; r < 8; ++r) { best[r] = 3.4e38f; bidx[r] = 0; }

  for (int it = 0; it < 64; ++it) {
    const int v0 = it * 128 + wave * 16;
    const unsigned short* crow = cbbf + (size_t)(v0 + n) * EDIM;
    if (it < 63) __builtin_prefetch(crow + 128 * EDIM, 0, 1);
    v8f acc{};
    #pragma unroll
    for (int s = 0; s < 8; ++s) {
      Frag Bf;   // B[k, n] = cb[v0+n, 32s + k], k = 8hi+{0..7}, 16+8hi+{0..7}
      const unsigned short* cr = crow + 32 * s + 8 * hi;
      Bf.q[0] = *(const uint4*)(cr);
      Bf.q[1] = *(const uint4*)(cr + 16);
      acc = wmma_bf16(A[s], Bf, acc);
    }
    float css = cbssq[v0 + n];
    #pragma unroll
    for (int r = 0; r < 8; ++r) {
      float d = __builtin_fmaf(-2.f, acc[r], css);
      if (d < best[r]) { best[r] = d; bidx[r] = v0 + n; }   // ascending v => first min kept
    }
  }

  // reduce across the 16 lanes sharing each row (stays within wave32 halves)
  #pragma unroll
  for (int mask = 8; mask >= 1; mask >>= 1) {
    #pragma unroll
    for (int r = 0; r < 8; ++r) {
      float ov = __shfl_xor(best[r], mask, 32);
      int   oi = __shfl_xor(bidx[r], mask, 32);
      if (ov < best[r] || (ov == best[r] && oi < bidx[r])) { best[r] = ov; bidx[r] = oi; }
    }
  }
  if (n == 0) {   // lanes 0 and 16 hold rows 0..7 / 8..15
    #pragma unroll
    for (int r = 0; r < 8; ++r) {
      rminS[wave][r + 8 * hi] = best[r];
      ridxS[wave][r + 8 * hi] = bidx[r];
    }
  }
  __syncthreads();
  if (tid < 16) {
    float bv = rminS[0][tid]; int bi = ridxS[0][tid];
    #pragma unroll
    for (int w = 1; w < 8; ++w) {
      float ov = rminS[w][tid]; int oi = ridxS[w][tid];
      if (ov < bv || (ov == bv && oi < bi)) { bv = ov; bi = oi; }
    }
    tok[p0 + tid] = bi;
    outTok[p0 + tid] = (float)bi;   // tokens appended to d_out as f32
  }
}

// ---------------------------------------------------------------------------
// Kernel 4: gather + post-quant GEMM  rec[p,c] = cb[token_p,:] . W_post[c,:] + b_post[c]
// Output scattered into [B,C,H,W]; per-lane stores contiguous in hw.
// ---------------------------------------------------------------------------
__global__ void k_postquant(const unsigned short* __restrict__ cbbf,
                            const float* __restrict__ Wpost,
                            const float* __restrict__ bpost,
                            const int* __restrict__ tok,
                            float* __restrict__ rec) {
  const int tid  = threadIdx.x;
  const int lane = tid & 31;
  const int wave = tid >> 5;
  const int hi   = lane >> 4;
  const int n    = lane & 15;
  const int p0   = blockIdx.x * 16;
  const int bb   = p0 >> 8;
  const int hw0  = p0 & 255;
  const int c0   = wave * 32;

  const int token = tok[p0 + n];
  const unsigned short* arow = cbbf + (size_t)token * EDIM;

  v8f acc0{}, acc1{};
  #pragma unroll
  for (int s = 0; s < 8; ++s) {
    Frag A;   // gathered codebook row, same fragment pattern as kernel 3
    const unsigned short* ar = arow + 32 * s + 8 * hi;
    A.q[0] = *(const uint4*)(ar);
    A.q[1] = *(const uint4*)(ar + 16);
    #pragma unroll
    for (int t = 0; t < 2; ++t) {
      const float* wrow = Wpost + (size_t)(c0 + 16 * t + n) * EDIM + 32 * s + 8 * hi;
      Frag Bf;
      #pragma unroll
      for (int j = 0; j < 8; ++j) {
        float2 f = *(const float2*)(wrow + 2 * (j & 3) + ((j >= 4) ? 16 : 0));
        Bf.u[j] = bf16pk(f.x, f.y);
      }
      if (t == 0) acc0 = wmma_bf16(A, Bf, acc0);
      else        acc1 = wmma_bf16(A, Bf, acc1);
    }
  }
  #pragma unroll
  for (int t = 0; t < 2; ++t) {
    int c = c0 + 16 * t + n;
    float bias = bpost[c];
    float* obase = rec + (size_t)bb * CDIM * HWD + (size_t)c * HWD + hw0;
    #pragma unroll
    for (int r = 0; r < 8; ++r) {
      obase[r + 8 * hi] = ((t == 0) ? acc0[r] : acc1[r]) + bias;
    }
  }
}

// ---------------------------------------------------------------------------
extern "C" void kernel_launch(void* const* d_in, const int* in_sizes, int n_in,
                              void* d_out, int out_size, void* d_ws, size_t ws_size,
                              hipStream_t stream) {
  const float* x     = (const float*)d_in[0];   // [B,C,H,W]
  const float* Wpre  = (const float*)d_in[1];   // [E,C]
  const float* bpre  = (const float*)d_in[2];   // [E]
  const float* cb    = (const float*)d_in[3];   // [V,E]
  const float* Wpost = (const float*)d_in[4];   // [C,E]
  const float* bpost = (const float*)d_in[5];   // [C]
  float* out = (float*)d_out;                   // rec (4194304 f32) ++ tokens (16384 f32)

  char* ws = (char*)d_ws;
  unsigned short* zbf   = (unsigned short*)(ws);             // 16384*256 bf16 = 8MB
  unsigned short* cbbf  = (unsigned short*)(ws + 8388608);   // 8192*256 bf16  = 4MB
  float*          cbssq = (float*)(ws + 12582912);           // 8192 f32
  int*            tokw  = (int*)(ws + 12615680);             // 16384 i32

  k_convert_codebook<<<VDIM / 8, 256, 0, stream>>>(cb, cbbf, cbssq);
  k_prequant       <<<NPIX / 16, 256, 0, stream>>>(x, Wpre, bpre, zbf);
  k_distargmin     <<<NPIX / 16, 256, 0, stream>>>(zbf, cbbf, cbssq, tokw, out + NREC);
  k_postquant      <<<NPIX / 16, 256, 0, stream>>>(cbbf, Wpost, bpost, tokw, out);
}
